// GatedMLP_58360015618262
// MI455X (gfx1250) — matrix operand order repaired and checked by reading, not compiled, not run on hardware
//
#include <hip/hip_runtime.h>
#include <hip/hip_bf16.h>
#include <stdint.h>

// ---------- types ----------
typedef __attribute__((ext_vector_type(8)))  unsigned short v8u;
typedef __attribute__((ext_vector_type(16))) unsigned short v16u;
typedef __attribute__((ext_vector_type(16))) __bf16         v16bf;
typedef __attribute__((ext_vector_type(8)))  float          v8f;
typedef int b128_t __attribute__((vector_size(16)));   // matches async-LDS builtin param

// ---------- problem dims (B*S=4096, H=4096, I=14336) ----------
#define TM 4096
#define TH 4096
#define TI 14336
#define BM 128
#define BN 128
#define BK 64
#define LDSP 72   // padded LDS row stride in bf16 elems (144B -> conflict-free)

// ---------- CDNA5 async global->LDS copy (direct, ASYNCcnt-tracked) ----------
#if __has_builtin(__builtin_amdgcn_global_load_async_to_lds_b128)
#define HAVE_ASYNC_LDS 1
static __device__ __forceinline__ void async_b128(const unsigned short* g,
                                                  unsigned short* l) {
  __builtin_amdgcn_global_load_async_to_lds_b128(
      (__attribute__((address_space(1))) b128_t*)(uintptr_t)g,
      (__attribute__((address_space(3))) b128_t*)(unsigned)(uintptr_t)l,
      0, 0);
}
static __device__ __forceinline__ void wait_async0() {
#if __has_builtin(__builtin_amdgcn_s_wait_asynccnt)
  __builtin_amdgcn_s_wait_asynccnt(0);
#else
  asm volatile("s_wait_asynccnt 0x0" ::: "memory");
#endif
}
#else
#define HAVE_ASYNC_LDS 0
#endif

static __device__ __forceinline__ unsigned short f2bf(float f) {
  unsigned int u = __builtin_bit_cast(unsigned int, f);
  unsigned int r = u + 0x7FFFu + ((u >> 16) & 1u);   // round-to-nearest-even
  return (unsigned short)(r >> 16);
}

static __device__ __forceinline__ v16bf make_frag(v8u lo, v8u hi) {
  v16u t = __builtin_shufflevector(lo, hi, 0,1,2,3,4,5,6,7,8,9,10,11,12,13,14,15);
  return __builtin_bit_cast(v16bf, t);
}

// ---------- conversion kernel: fp32 -> bf16, 8 elems/thread ----------
__global__ __launch_bounds__(256)
void GatedMLP_cvt_bf16(const float* __restrict__ in, unsigned short* __restrict__ out) {
  size_t i = ((size_t)blockIdx.x * 256 + threadIdx.x) * 8;
  const float4* p = (const float4*)(in + i);
  float4 a = p[0], b = p[1];
  union { unsigned short s[8]; v8u v; } t;
  t.s[0] = f2bf(a.x); t.s[1] = f2bf(a.y); t.s[2] = f2bf(a.z); t.s[3] = f2bf(a.w);
  t.s[4] = f2bf(b.x); t.s[5] = f2bf(b.y); t.s[6] = f2bf(b.z); t.s[7] = f2bf(b.w);
  *(v8u*)(out + i) = t.v;
}

// ============================================================================
//  FAST PATH: all operands bf16 in memory; every tile staged via async copies
// ============================================================================

// stage one 128x64 bf16 tile: 4 x b128 per thread (256 threads)
static __device__ __forceinline__ void stage_tile(const unsigned short* __restrict__ gsrc,
                                                  size_t gstride, int k0,
                                                  unsigned short* __restrict__ lds,
                                                  int cA, int rA
#if !HAVE_ASYNC_LDS
                                                  , v8u* regs
#endif
                                                  ) {
#pragma unroll
  for (int p = 0; p < 4; ++p) {
    const int r = rA + p * 32;
#if HAVE_ASYNC_LDS
    async_b128(gsrc + (size_t)r * gstride + k0 + cA, &lds[r * LDSP + cA]);
#else
    regs[p] = *(const v8u*)(gsrc + (size_t)r * gstride + k0 + cA);
#endif
  }
#if HAVE_ASYNC_LDS
  (void)cA; (void)rA;
#endif
}

// ---------- fused gate+up GEMM + silu*mul (bf16 weights) ----------
__global__ __launch_bounds__(256)
void GatedMLP_gate_up_bf(const unsigned short* __restrict__ xb,
                         const unsigned short* __restrict__ wgb,
                         const unsigned short* __restrict__ wub,
                         unsigned short* __restrict__ a_out) {
  __shared__ unsigned short ldsA[2][BM * LDSP];
  __shared__ unsigned short ldsG[2][BN * LDSP];
  __shared__ unsigned short ldsU[2][BN * LDSP];

  const int tid  = threadIdx.x;
  const int lane = tid & 31;
  const int wave = tid >> 5;
  const int wm   = wave & 1;
  const int wn   = wave >> 1;
  const int m0   = blockIdx.x * BM;
  const int n0   = blockIdx.y * BN;
  const int l15  = lane & 15;
  const int lh   = lane >> 4;
  const int cA   = (tid & 7) * 8;
  const int rA   = tid >> 3;

  const unsigned short* gA = xb  + (size_t)m0 * TH;
  const unsigned short* gG = wgb + (size_t)n0 * TH;
  const unsigned short* gU = wub + (size_t)n0 * TH;

  v8f accG[4][2], accU[4][2];
#pragma unroll
  for (int i = 0; i < 4; ++i)
#pragma unroll
    for (int j = 0; j < 2; ++j) { accG[i][j] = {}; accU[i][j] = {}; }

#if !HAVE_ASYNC_LDS
  v8u ra[4], rg[4], ru[4];
#define STAGE(src, st, k0, lds, regs) stage_tile(src, st, k0, lds, cA, rA, regs)
#define FLUSH(lds, regs)                                             \
  { _Pragma("unroll") for (int p = 0; p < 4; ++p)                    \
      *(v8u*)(&(lds)[(rA + p * 32) * LDSP + cA]) = (regs)[p]; }
#else
#define STAGE(src, st, k0, lds, regs) stage_tile(src, st, k0, lds, cA, rA)
#define FLUSH(lds, regs)
#endif

  // prologue: fill buffer 0
  STAGE(gA, TH, 0, ldsA[0], ra);
  STAGE(gG, TH, 0, ldsG[0], rg);
  STAGE(gU, TH, 0, ldsU[0], ru);
  FLUSH(ldsA[0], ra); FLUSH(ldsG[0], rg); FLUSH(ldsU[0], ru);
#if HAVE_ASYNC_LDS
  wait_async0();
#endif
  __syncthreads();

  const int T = TH / BK;   // 64
  for (int t = 0; t < T; ++t) {
    const int cur = t & 1;
    const int nxt = cur ^ 1;
    const int k1  = (t + 1) * BK;
    const bool more = (t + 1) < T;

    if (more) {
      STAGE(gA, TH, k1, ldsA[nxt], ra);
      STAGE(gG, TH, k1, ldsG[nxt], rg);
      STAGE(gU, TH, k1, ldsU[nxt], ru);
    }

#pragma unroll
    for (int kk = 0; kk < BK; kk += 32) {
      v16bf af[4];
#pragma unroll
      for (int i = 0; i < 4; ++i) {
        const int row = wm * 64 + i * 16 + l15;
        const unsigned short* b = &ldsA[cur][row * LDSP + kk + lh * 16];
        af[i] = make_frag(*(const v8u*)b, *(const v8u*)(b + 8));
      }
      v16bf bg[2], bu[2];
#pragma unroll
      for (int j = 0; j < 2; ++j) {
        const int row = wn * 32 + j * 16 + l15;
        const unsigned short* pg = &ldsG[cur][row * LDSP + kk + lh * 16];
        const unsigned short* pu = &ldsU[cur][row * LDSP + kk + lh * 16];
        bg[j] = make_frag(*(const v8u*)pg, *(const v8u*)(pg + 8));
        bu[j] = make_frag(*(const v8u*)pu, *(const v8u*)(pu + 8));
      }
#pragma unroll
      for (int i = 0; i < 4; ++i)
#pragma unroll
        for (int j = 0; j < 2; ++j) {
          accG[i][j] = __builtin_amdgcn_wmma_f32_16x16x32_bf16(
              false, af[i], false, bg[j], (short)0, accG[i][j], false, false);
          accU[i][j] = __builtin_amdgcn_wmma_f32_16x16x32_bf16(
              false, af[i], false, bu[j], (short)0, accU[i][j], false, false);
        }
    }

    if (more) { FLUSH(ldsA[nxt], ra); FLUSH(ldsG[nxt], rg); FLUSH(ldsU[nxt], ru); }
#if HAVE_ASYNC_LDS
    wait_async0();
#endif
    __syncthreads();
  }
#undef STAGE
#undef FLUSH

#pragma unroll
  for (int i = 0; i < 4; ++i)
#pragma unroll
    for (int j = 0; j < 2; ++j) {
      const int col  = n0 + wn * 32 + j * 16 + l15;
      const int rowb = m0 + wm * 64 + i * 16 + lh * 8;
#pragma unroll
      for (int v = 0; v < 8; ++v) {
        const float g = accG[i][j][v];
        const float u = accU[i][j][v];
        const float h = (g / (1.0f + __expf(-g))) * u;
        a_out[(size_t)(rowb + v) * TI + col] = f2bf(h);
      }
    }
}

// ---------- down GEMM (bf16 weights): d = a @ Wd^T ----------
__global__ __launch_bounds__(256)
void GatedMLP_down_bf(const unsigned short* __restrict__ ab,
                      const unsigned short* __restrict__ wdb,
                      float* __restrict__ out) {
  __shared__ unsigned short ldsA[2][BM * LDSP];
  __shared__ unsigned short ldsB[2][BN * LDSP];

  const int tid  = threadIdx.x;
  const int lane = tid & 31;
  const int wave = tid >> 5;
  const int wm   = wave & 1;
  const int wn   = wave >> 1;
  const int m0   = blockIdx.x * BM;
  const int n0   = blockIdx.y * BN;
  const int l15  = lane & 15;
  const int lh   = lane >> 4;
  const int cA   = (tid & 7) * 8;
  const int rA   = tid >> 3;

  const unsigned short* gA = ab  + (size_t)m0 * TI;
  const unsigned short* gB = wdb + (size_t)n0 * TI;

  v8f acc[4][2];
#pragma unroll
  for (int i = 0; i < 4; ++i)
#pragma unroll
    for (int j = 0; j < 2; ++j) acc[i][j] = {};

#if !HAVE_ASYNC_LDS
  v8u ra[4], rb[4];
#define STAGE(src, st, k0, lds, regs) stage_tile(src, st, k0, lds, cA, rA, regs)
#define FLUSH(lds, regs)                                             \
  { _Pragma("unroll") for (int p = 0; p < 4; ++p)                    \
      *(v8u*)(&(lds)[(rA + p * 32) * LDSP + cA]) = (regs)[p]; }
#else
#define STAGE(src, st, k0, lds, regs) stage_tile(src, st, k0, lds, cA, rA)
#define FLUSH(lds, regs)
#endif

  STAGE(gA, TI, 0, ldsA[0], ra);
  STAGE(gB, TI, 0, ldsB[0], rb);
  FLUSH(ldsA[0], ra); FLUSH(ldsB[0], rb);
#if HAVE_ASYNC_LDS
  wait_async0();
#endif
  __syncthreads();

  const int T = TI / BK;   // 224
  for (int t = 0; t < T; ++t) {
    const int cur = t & 1;
    const int nxt = cur ^ 1;
    const int k1  = (t + 1) * BK;
    const bool more = (t + 1) < T;

    if (more) {
      STAGE(gA, TI, k1, ldsA[nxt], ra);
      STAGE(gB, TI, k1, ldsB[nxt], rb);
    }

#pragma unroll
    for (int kk = 0; kk < BK; kk += 32) {
      v16bf af[4];
#pragma unroll
      for (int i = 0; i < 4; ++i) {
        const int row = wm * 64 + i * 16 + l15;
        const unsigned short* b = &ldsA[cur][row * LDSP + kk + lh * 16];
        af[i] = make_frag(*(const v8u*)b, *(const v8u*)(b + 8));
      }
      v16bf bf[2];
#pragma unroll
      for (int j = 0; j < 2; ++j) {
        const int row = wn * 32 + j * 16 + l15;
        const unsigned short* pb = &ldsB[cur][row * LDSP + kk + lh * 16];
        bf[j] = make_frag(*(const v8u*)pb, *(const v8u*)(pb + 8));
      }
#pragma unroll
      for (int i = 0; i < 4; ++i)
#pragma unroll
        for (int j = 0; j < 2; ++j)
          acc[i][j] = __builtin_amdgcn_wmma_f32_16x16x32_bf16(
              false, af[i], false, bf[j], (short)0, acc[i][j], false, false);
    }

    if (more) { FLUSH(ldsA[nxt], ra); FLUSH(ldsB[nxt], rb); }
#if HAVE_ASYNC_LDS
    wait_async0();
#endif
    __syncthreads();
  }
#undef STAGE
#undef FLUSH

#pragma unroll
  for (int i = 0; i < 4; ++i)
#pragma unroll
    for (int j = 0; j < 2; ++j) {
      const int col  = n0 + wn * 32 + j * 16 + l15;
      const int rowb = m0 + wm * 64 + i * 16 + lh * 8;
#pragma unroll
      for (int v = 0; v < 8; ++v)
        out[(size_t)(rowb + v) * TH + col] = acc[i][j][v];
    }
}

// ============================================================================
//  FALLBACK PATH (small workspace): fp32 weights converted on the fly
// ============================================================================
__global__ __launch_bounds__(256)
void GatedMLP_gate_up_f32w(const unsigned short* __restrict__ xb,
                           const float* __restrict__ wg,
                           const float* __restrict__ wu,
                           unsigned short* __restrict__ a_out) {
  __shared__ unsigned short ldsA[2][BM * LDSP];
  __shared__ unsigned short ldsG[2][BN * LDSP];
  __shared__ unsigned short ldsU[2][BN * LDSP];

  const int tid  = threadIdx.x;
  const int lane = tid & 31;
  const int wave = tid >> 5;
  const int wm   = wave & 1;
  const int wn   = wave >> 1;
  const int m0   = blockIdx.x * BM;
  const int n0   = blockIdx.y * BN;
  const int l15  = lane & 15;
  const int lh   = lane >> 4;
  const int cA   = (tid & 7) * 8;
  const int rA   = tid >> 3;
  const int cW   = (tid & 3) * 16;
  const int rW   = tid >> 2;

  v8f accG[4][2], accU[4][2];
#pragma unroll
  for (int i = 0; i < 4; ++i)
#pragma unroll
    for (int j = 0; j < 2; ++j) { accG[i][j] = {}; accU[i][j] = {}; }

  float4 rg[8], ru[8];
#if !HAVE_ASYNC_LDS
  v8u ra[4];
#endif

  auto issueA = [&](int k0, int buf) {
#pragma unroll
    for (int p = 0; p < 4; ++p) {
      const int r = rA + p * 32;
#if HAVE_ASYNC_LDS
      async_b128(xb + (size_t)(m0 + r) * TH + k0 + cA, &ldsA[buf][r * LDSP + cA]);
#else
      ra[p] = *(const v8u*)(xb + (size_t)(m0 + r) * TH + k0 + cA);
#endif
    }
  };
  auto storeA = [&](int buf) {
#if !HAVE_ASYNC_LDS
#pragma unroll
    for (int p = 0; p < 4; ++p)
      *(v8u*)(&ldsA[buf][(rA + p * 32) * LDSP + cA]) = ra[p];
#endif
    (void)buf;
  };
  auto issueW = [&](int k0) {
#pragma unroll
    for (int p = 0; p < 2; ++p) {
      const float4* sg = (const float4*)(wg + (size_t)(n0 + rW + p * 64) * TH + k0 + cW);
      const float4* su = (const float4*)(wu + (size_t)(n0 + rW + p * 64) * TH + k0 + cW);
#pragma unroll
      for (int q = 0; q < 4; ++q) { rg[p * 4 + q] = sg[q]; ru[p * 4 + q] = su[q]; }
    }
  };
  auto storeW = [&](int buf) {
#pragma unroll
    for (int p = 0; p < 2; ++p) {
      const int r = rW + p * 64;
      union { unsigned short s[16]; v8u v[2]; } tg, tu;
#pragma unroll
      for (int q = 0; q < 4; ++q) {
        float4 fg = rg[p * 4 + q], fu = ru[p * 4 + q];
        tg.s[4*q+0] = f2bf(fg.x); tg.s[4*q+1] = f2bf(fg.y);
        tg.s[4*q+2] = f2bf(fg.z); tg.s[4*q+3] = f2bf(fg.w);
        tu.s[4*q+0] = f2bf(fu.x); tu.s[4*q+1] = f2bf(fu.y);
        tu.s[4*q+2] = f2bf(fu.z); tu.s[4*q+3] = f2bf(fu.w);
      }
      *(v8u*)(&ldsG[buf][r * LDSP + cW])     = tg.v[0];
      *(v8u*)(&ldsG[buf][r * LDSP + cW + 8]) = tg.v[1];
      *(v8u*)(&ldsU[buf][r * LDSP + cW])     = tu.v[0];
      *(v8u*)(&ldsU[buf][r * LDSP + cW + 8]) = tu.v[1];
    }
  };

  issueA(0, 0); issueW(0); storeA(0); storeW(0);
#if HAVE_ASYNC_LDS
  wait_async0();
#endif
  __syncthreads();

  const int T = TH / BK;
  for (int t = 0; t < T; ++t) {
    const int cur = t & 1, nxt = cur ^ 1;
    const int k1 = (t + 1) * BK;
    const bool more = (t + 1) < T;
    if (more) { issueA(k1, nxt); issueW(k1); }

#pragma unroll
    for (int kk = 0; kk < BK; kk += 32) {
      v16bf af[4];
#pragma unroll
      for (int i = 0; i < 4; ++i) {
        const int row = wm * 64 + i * 16 + l15;
        const unsigned short* b = &ldsA[cur][row * LDSP + kk + lh * 16];
        af[i] = make_frag(*(const v8u*)b, *(const v8u*)(b + 8));
      }
      v16bf bg[2], bu[2];
#pragma unroll
      for (int j = 0; j < 2; ++j) {
        const int row = wn * 32 + j * 16 + l15;
        const unsigned short* pg = &ldsG[cur][row * LDSP + kk + lh * 16];
        const unsigned short* pu = &ldsU[cur][row * LDSP + kk + lh * 16];
        bg[j] = make_frag(*(const v8u*)pg, *(const v8u*)(pg + 8));
        bu[j] = make_frag(*(const v8u*)pu, *(const v8u*)(pu + 8));
      }
#pragma unroll
      for (int i = 0; i < 4; ++i)
#pragma unroll
        for (int j = 0; j < 2; ++j) {
          accG[i][j] = __builtin_amdgcn_wmma_f32_16x16x32_bf16(
              false, af[i], false, bg[j], (short)0, accG[i][j], false, false);
          accU[i][j] = __builtin_amdgcn_wmma_f32_16x16x32_bf16(
              false, af[i], false, bu[j], (short)0, accU[i][j], false, false);
        }
    }

    if (more) { storeA(nxt); storeW(nxt); }
#if HAVE_ASYNC_LDS
    wait_async0();
#endif
    __syncthreads();
  }

#pragma unroll
  for (int i = 0; i < 4; ++i)
#pragma unroll
    for (int j = 0; j < 2; ++j) {
      const int col  = n0 + wn * 32 + j * 16 + l15;
      const int rowb = m0 + wm * 64 + i * 16 + lh * 8;
#pragma unroll
      for (int v = 0; v < 8; ++v) {
        const float g = accG[i][j][v];
        const float u = accU[i][j][v];
        const float h = (g / (1.0f + __expf(-g))) * u;
        a_out[(size_t)(rowb + v) * TI + col] = f2bf(h);
      }
    }
}

__global__ __launch_bounds__(256)
void GatedMLP_down_f32w(const unsigned short* __restrict__ ab,
                        const float* __restrict__ wd,
                        float* __restrict__ out) {
  __shared__ unsigned short ldsA[2][BM * LDSP];
  __shared__ unsigned short ldsB[2][BN * LDSP];

  const int tid  = threadIdx.x;
  const int lane = tid & 31;
  const int wave = tid >> 5;
  const int wm   = wave & 1;
  const int wn   = wave >> 1;
  const int m0   = blockIdx.x * BM;
  const int n0   = blockIdx.y * BN;
  const int l15  = lane & 15;
  const int lh   = lane >> 4;
  const int cA   = (tid & 7) * 8;
  const int rA   = tid >> 3;
  const int cW   = (tid & 3) * 16;
  const int rW   = tid >> 2;

  v8f acc[4][2];
#pragma unroll
  for (int i = 0; i < 4; ++i)
#pragma unroll
    for (int j = 0; j < 2; ++j) acc[i][j] = {};

  float4 rb[8];
#if !HAVE_ASYNC_LDS
  v8u ra[4];
#endif

  auto issueA = [&](int k0, int buf) {
#pragma unroll
    for (int p = 0; p < 4; ++p) {
      const int r = rA + p * 32;
#if HAVE_ASYNC_LDS
      async_b128(ab + (size_t)(m0 + r) * TI + k0 + cA, &ldsA[buf][r * LDSP + cA]);
#else
      ra[p] = *(const v8u*)(ab + (size_t)(m0 + r) * TI + k0 + cA);
#endif
    }
  };
  auto storeA = [&](int buf) {
#if !HAVE_ASYNC_LDS
#pragma unroll
    for (int p = 0; p < 4; ++p)
      *(v8u*)(&ldsA[buf][(rA + p * 32) * LDSP + cA]) = ra[p];
#endif
    (void)buf;
  };
  auto issueW = [&](int k0) {
#pragma unroll
    for (int p = 0; p < 2; ++p) {
      const float4* s4 = (const float4*)(wd + (size_t)(n0 + rW + p * 64) * TI + k0 + cW);
#pragma unroll
      for (int q = 0; q < 4; ++q) rb[p * 4 + q] = s4[q];
    }
  };
  auto storeW = [&](int buf) {
#pragma unroll
    for (int p = 0; p < 2; ++p) {
      const int r = rW + p * 64;
      union { unsigned short s[16]; v8u v[2]; } tb;
#pragma unroll
      for (int q = 0; q < 4; ++q) {
        float4 f = rb[p * 4 + q];
        tb.s[4*q+0] = f2bf(f.x); tb.s[4*q+1] = f2bf(f.y);
        tb.s[4*q+2] = f2bf(f.z); tb.s[4*q+3] = f2bf(f.w);
      }
      *(v8u*)(&ldsB[buf][r * LDSP + cW])     = tb.v[0];
      *(v8u*)(&ldsB[buf][r * LDSP + cW + 8]) = tb.v[1];
    }
  };

  issueA(0, 0); issueW(0); storeA(0); storeW(0);
#if HAVE_ASYNC_LDS
  wait_async0();
#endif
  __syncthreads();

  const int T = TI / BK;
  for (int t = 0; t < T; ++t) {
    const int cur = t & 1, nxt = cur ^ 1;
    const int k1 = (t + 1) * BK;
    const bool more = (t + 1) < T;
    if (more) { issueA(k1, nxt); issueW(k1); }

#pragma unroll
    for (int kk = 0; kk < BK; kk += 32) {
      v16bf af[4];
#pragma unroll
      for (int i = 0; i < 4; ++i) {
        const int row = wm * 64 + i * 16 + l15;
        const unsigned short* b = &ldsA[cur][row * LDSP + kk + lh * 16];
        af[i] = make_frag(*(const v8u*)b, *(const v8u*)(b + 8));
      }
      v16bf bf[2];
#pragma unroll
      for (int j = 0; j < 2; ++j) {
        const int row = wn * 32 + j * 16 + l15;
        const unsigned short* pb = &ldsB[cur][row * LDSP + kk + lh * 16];
        bf[j] = make_frag(*(const v8u*)pb, *(const v8u*)(pb + 8));
      }
#pragma unroll
      for (int i = 0; i < 4; ++i)
#pragma unroll
        for (int j = 0; j < 2; ++j)
          acc[i][j] = __builtin_amdgcn_wmma_f32_16x16x32_bf16(
              false, af[i], false, bf[j], (short)0, acc[i][j], false, false);
    }

    if (more) { storeA(nxt); storeW(nxt); }
#if HAVE_ASYNC_LDS
    wait_async0();
#endif
    __syncthreads();
  }

#pragma unroll
  for (int i = 0; i < 4; ++i)
#pragma unroll
    for (int j = 0; j < 2; ++j) {
      const int col  = n0 + wn * 32 + j * 16 + l15;
      const int rowb = m0 + wm * 64 + i * 16 + lh * 8;
#pragma unroll
      for (int v = 0; v < 8; ++v)
        out[(size_t)(rowb + v) * TH + col] = acc[i][j][v];
    }
}

// ---------- host ----------
extern "C" void kernel_launch(void* const* d_in, const int* in_sizes, int n_in,
                              void* d_out, int out_size, void* d_ws, size_t ws_size,
                              hipStream_t stream) {
  (void)in_sizes; (void)n_in; (void)out_size;
  const float* x  = (const float*)d_in[0];
  const float* wg = (const float*)d_in[1];
  const float* wu = (const float*)d_in[2];
  const float* wd = (const float*)d_in[3];
  float* out = (float*)d_out;

  const size_t A_B  = (size_t)TM * TI * 2;   // a bf16
  const size_t X_B  = (size_t)TM * TH * 2;   // x bf16
  const size_t W_B  = (size_t)TI * TH * 2;   // each weight bf16
  const size_t FULL = A_B + X_B + 3 * W_B;   // 480 MB

  char* w = (char*)d_ws;
  unsigned short* a_bf = (unsigned short*)w;            w += A_B;
  unsigned short* x_bf = (unsigned short*)w;            w += X_B;

  const unsigned xcv = (unsigned)(((size_t)TM * TH) / 2048);  // 8192
  const unsigned wcv = (unsigned)(((size_t)TI * TH) / 2048);  // 28672

  if (ws_size >= FULL) {
    unsigned short* wg_bf = (unsigned short*)w;         w += W_B;
    unsigned short* wu_bf = (unsigned short*)w;         w += W_B;
    unsigned short* wd_bf = (unsigned short*)w;
    GatedMLP_cvt_bf16<<<xcv, 256, 0, stream>>>(x,  x_bf);
    GatedMLP_cvt_bf16<<<wcv, 256, 0, stream>>>(wg, wg_bf);
    GatedMLP_cvt_bf16<<<wcv, 256, 0, stream>>>(wu, wu_bf);
    GatedMLP_cvt_bf16<<<wcv, 256, 0, stream>>>(wd, wd_bf);
    GatedMLP_gate_up_bf<<<dim3(TM / BM, TI / BN), 256, 0, stream>>>(x_bf, wg_bf, wu_bf, a_bf);
    GatedMLP_down_bf<<<dim3(TM / BM, TH / BN), 256, 0, stream>>>(a_bf, wd_bf, out);
  } else {
    GatedMLP_cvt_bf16<<<xcv, 256, 0, stream>>>(x, x_bf);
    GatedMLP_gate_up_f32w<<<dim3(TM / BM, TI / BN), 256, 0, stream>>>(x_bf, wg, wu, a_bf);
    GatedMLP_down_f32w<<<dim3(TM / BM, TH / BN), 256, 0, stream>>>(a_bf, wd, out);
  }
}
